// R_SCNN_17918603559438
// MI455X (gfx1250) — compile-verified, matrix operand order checked by reading, not run on hardware
//
#include <hip/hip_runtime.h>
#include <math.h>

typedef __bf16 bf16;
typedef __attribute__((ext_vector_type(16))) bf16  v16bf;
typedef __attribute__((ext_vector_type(8)))  bf16  bf16x8;
typedef __attribute__((ext_vector_type(8)))  float v8f;

#define NB 8
#define NC 256
#define NH 256
#define NW 25
#define NS 2048           /* NB*NH */
#define HH2 512
#define NCONV 102400      /* NB*HH2*NW */
#define KC 2304           /* NC*9 */
#define LDS_STRIDE 40     /* bf16 elems per staged column (32 + pad, bank-friendly) */

// Load a 16-element bf16 WMMA fragment (A row-per-lane / B col-per-lane layout):
// elements 0..7  <- 8 contiguous bf16 at p       (K = k0 + hi8 + e)
// elements 8..15 <- 8 contiguous bf16 at p + 16  (K = k0 + 16 + hi8 + e)
__device__ __forceinline__ v16bf load_frag16(const bf16* p) {
  bf16x8 lo = *(const bf16x8*)p;
  bf16x8 hi = *(const bf16x8*)(p + 16);
  v16bf f;
#pragma unroll
  for (int e = 0; e < 8; ++e) { f[e] = lo[e]; f[e + 8] = hi[e]; }
  return f;
}

// ---------- weight conversion + BN constants ----------
__global__ __launch_bounds__(256) void prep_kernel(
    const float* __restrict__ Wmsg, const float* __restrict__ Wup2,
    const float* __restrict__ gamma, const float* __restrict__ beta,
    const float* __restrict__ mean, const float* __restrict__ var,
    bf16* __restrict__ Wc, bf16* __restrict__ Wb,
    float* __restrict__ bnA, float* __restrict__ bnB) {
  const int tid = blockIdx.x * 256 + threadIdx.x;
  if (tid < 256) {
    const float inv = gamma[tid] / sqrtf(var[tid] + 1e-5f);
    bnA[tid] = inv;
    bnB[tid] = beta[tid] - mean[tid] * inv;
  }
  if (tid < 65536) {  // Wc[o][c] = W_msg[o,c,0,4]
    const int o = tid >> 8, c = tid & 255;
    Wc[tid] = (bf16)Wmsg[o * 2304 + c * 9 + 4];
  }
  if (tid < 256 * KC) {  // Wb[o][(kh*3+kw)*256 + i] = W_up2[o,i,kh,kw]
    const int o = tid / KC, kk = tid % KC;
    const int r = kk >> 8, i = kk & 255;
    Wb[tid] = (bf16)Wup2[o * 2304 + i * 9 + r];
  }
}

// ---------- scan initialization (slice 0 of the scan -> slot 24) ----------
__global__ __launch_bounds__(256) void scan_init_kernel(
    const float* __restrict__ src, int mode, float* __restrict__ xdst,
    bf16* __restrict__ cdst, const float* __restrict__ bnA,
    const float* __restrict__ bnB) {
  const int tid = blockIdx.x * 256 + threadIdx.x;  // 524288 threads
  const int n = tid >> 8, c = tid & 255;
  float v;
  if (mode == 0) {  // gather p2_r[b,c,h,w=0]
    const int b = n >> 8, h = n & 255;
    v = src[(((size_t)b * NC + c) * NH + h) * NW + 0];
  } else {          // X1 slot 24, layout (slot,n,c)
    v = src[((size_t)24 * NS + n) * NC + c];
  }
  cdst[tid] = (bf16)v;
  const float y = v * bnA[c] + bnB[c];
  xdst[((size_t)24 * NS + n) * NC + c] = y > 0.f ? y : 0.f;
}

// ---------- one recurrence step: new = xi + relu(Wc @ carry) ----------
__global__ __launch_bounds__(256) void scan_step_kernel(
    const bf16* __restrict__ Wc, const bf16* __restrict__ cin,
    const float* __restrict__ xsrc, float* __restrict__ xout,
    bf16* __restrict__ cout_, const float* __restrict__ bnA,
    const float* __restrict__ bnB, int w_in, int slot_out, int mode) {
  const int gw   = (blockIdx.x * 256 + threadIdx.x) >> 5;  // 0..2047
  const int lane = threadIdx.x & 31;
  const int m0 = (gw & 15) << 4;
  const int n0 = (gw >> 4) << 4;
  const int ml = lane & 15;
  const int hi8 = (lane >> 4) << 3;

  v8f acc;
#pragma unroll
  for (int i = 0; i < 8; ++i) acc[i] = 0.f;

  const bf16* pa = Wc  + (size_t)(m0 + ml) * NC + hi8;
  const bf16* pb = cin + (size_t)(n0 + ml) * NC + hi8;
#pragma unroll
  for (int k0 = 0; k0 < NC; k0 += 32) {
    v16bf a = load_frag16(pa + k0);
    v16bf b = load_frag16(pb + k0);
    acc = __builtin_amdgcn_wmma_f32_16x16x32_bf16(false, a, false, b,
                                                  (short)0, acc, false, false);
  }

  const int n = n0 + ml;
  const int bb = n >> 8, h = n & 255;
#pragma unroll
  for (int v = 0; v < 8; ++v) {
    const int o = m0 + hi8 + v;
    float g = acc[v];
    g = g > 0.f ? g : 0.f;
    float xi;
    if (mode == 0)
      xi = xsrc[(((size_t)bb * NC + o) * NH + h) * NW + w_in];  // p2_r slice
    else
      xi = xsrc[((size_t)w_in * NS + n) * NC + o];              // X1 slot
    const float nv = xi + g;
    cout_[(size_t)n * NC + o] = (bf16)nv;           // raw carry for next step
    const float y = nv * bnA[o] + bnB[o];           // fused BN+ReLU
    xout[((size_t)slot_out * NS + n) * NC + o] = y > 0.f ? y : 0.f;
  }
}

// ---------- upsample H 256->512 and convert to bf16, layout (b,h2,w,c) ----------
__global__ __launch_bounds__(256) void upsample_bf16_kernel(
    const float* __restrict__ X2, bf16* __restrict__ U) {
  const int tid = blockIdx.x * 256 + threadIdx.x;
  const int c = tid & 255;
  int t = tid >> 8;
  const int w = t % 25; t /= 25;
  const int h2 = t % HH2;
  const int b = t / HH2;
  const float pos = h2 * (255.0f / 511.0f);
  const int i0 = (int)pos;
  const float fr = pos - (float)i0;
  int i1 = i0 + 1; if (i1 > 255) i1 = 255;
  const float v0 = X2[((size_t)w * NS + (size_t)b * NH + i0) * NC + c];
  const float v1 = X2[((size_t)w * NS + (size_t)b * NH + i1) * NC + c];
  U[tid] = (bf16)(v0 + (v1 - v0) * fr);
}

// ---------- 3x3 conv 256->256 as LDS-staged implicit GEMM ----------
// Block: 8 waves cover all M=256 (wave w -> rows [32w,32w+32)) x 64 columns.
// Im2col K-chunk (32 x 64 cols) double-buffered in LDS; B read from global once
// per block instead of once per M-tile (16x less L2 traffic).
__global__ __launch_bounds__(256) void conv3_kernel(
    const bf16* __restrict__ Wb, const bf16* __restrict__ U,
    bf16* __restrict__ Cout) {
  __shared__ bf16 lds[2][64 * LDS_STRIDE];   // 10 KB

  const int tid  = threadIdx.x;
  const int wave = tid >> 5;
  const int lane = tid & 31;
  const int n0 = blockIdx.x * 64;
  const int m0 = wave * 32;
  const int ml = lane & 15;
  const int hi8 = (lane >> 4) << 3;

  // staging role: thread stages column jcol, 8-element k-quarter q
  const int jcol = tid >> 2;
  const int q    = tid & 3;
  const int nn = n0 + jcol;
  const int scw = nn % 25;
  const int st  = nn / 25;
  const int sch = st & (HH2 - 1);
  const int scb = st >> 9;

  v8f acc[2][4];
#pragma unroll
  for (int mt = 0; mt < 2; ++mt)
#pragma unroll
    for (int j = 0; j < 4; ++j)
#pragma unroll
      for (int i = 0; i < 8; ++i) acc[mt][j][i] = 0.f;

  const bf16* pa0 = Wb + (size_t)(m0 + ml) * KC + hi8;
  const bf16* pa1 = Wb + (size_t)(m0 + 16 + ml) * KC + hi8;

  auto stage = [&](int k0, int buf) {
    const int r  = k0 >> 8;                 // tap 0..8 (chunks never cross taps)
    const int kh = r / 3, kw = r - kh * 3;
    const int i  = (k0 & 255) + q * 8;
    const int hh = sch + kh - 1;
    const int ww = scw + kw - 1;
    bf16x8 v;
    if (hh >= 0 && hh < HH2 && ww >= 0 && ww < 25) {
      v = *(const bf16x8*)(U + (((size_t)(scb * HH2 + hh) * 25 + ww) << 8) + i);
    } else {
#pragma unroll
      for (int e = 0; e < 8; ++e) v[e] = (bf16)0.0f;
    }
    *(bf16x8*)(&lds[buf][jcol * LDS_STRIDE + q * 8]) = v;
  };

  stage(0, 0);
  __syncthreads();

  for (int t = 0; t < 72; ++t) {
    const int k0 = t * 32;
    if (t + 1 < 72) stage(k0 + 32, (t + 1) & 1);
    __builtin_prefetch(pa0 + k0 + 32, 0, 3);   // next weight chunk
    const bf16* lbuf = lds[t & 1];
    v16bf a0 = load_frag16(pa0 + k0);
    v16bf a1 = load_frag16(pa1 + k0);
#pragma unroll
    for (int j = 0; j < 4; ++j) {
      v16bf b = load_frag16(lbuf + (j * 16 + ml) * LDS_STRIDE + hi8);
      acc[0][j] = __builtin_amdgcn_wmma_f32_16x16x32_bf16(false, a0, false, b,
                                                          (short)0, acc[0][j], false, false);
      acc[1][j] = __builtin_amdgcn_wmma_f32_16x16x32_bf16(false, a1, false, b,
                                                          (short)0, acc[1][j], false, false);
    }
    __syncthreads();
  }

#pragma unroll
  for (int mt = 0; mt < 2; ++mt)
#pragma unroll
    for (int j = 0; j < 4; ++j) {
      const int n = n0 + j * 16 + ml;
      const int o = m0 + mt * 16 + hi8;
      bf16x8 s;
#pragma unroll
      for (int v = 0; v < 8; ++v) s[v] = (bf16)acc[mt][j][v];
      *(bf16x8*)(Cout + ((size_t)n << 8) + o) = s;
    }
}

// ---------- 256->1 channel dot: z[b,h2,w]; one wave per output ----------
__global__ __launch_bounds__(256) void chan_reduce_kernel(
    const bf16* __restrict__ Cout, const float* __restrict__ Wconv,
    float* __restrict__ z) {
  const int gw   = (blockIdx.x * 256 + threadIdx.x) >> 5;  // n index < 102400
  const int lane = threadIdx.x & 31;
  bf16x8 vv = *(const bf16x8*)(Cout + (size_t)gw * NC + lane * 8);
  float acc = 0.f;
#pragma unroll
  for (int j = 0; j < 8; ++j) acc += (float)vv[j] * Wconv[lane * 8 + j];
#pragma unroll
  for (int off = 16; off > 0; off >>= 1) acc += __shfl_down(acc, off, 32);
  if (lane == 0) z[gw] = acc;
}

// ---------- two upsamples + 25->1 dot + sigmoid ----------
__global__ __launch_bounds__(256) void tail_kernel(
    const float* __restrict__ z, const float* __restrict__ Wconv2,
    float* __restrict__ out) {
  const int tid = blockIdx.x * 256 + threadIdx.x;  // 16384
  const int b = tid >> 11;
  const int h4 = tid & 2047;
  const float pos3 = h4 * (1023.0f / 2047.0f);
  const int h3a = (int)pos3;
  const float t3 = pos3 - (float)h3a;
  int h3b = h3a + 1; if (h3b > 1023) h3b = 1023;

  int rows[4]; float cw[4];
  const int h3s[2] = {h3a, h3b};
  const float tw[2] = {1.f - t3, t3};
#pragma unroll
  for (int s = 0; s < 2; ++s) {
    const float pos2 = h3s[s] * (511.0f / 1023.0f);
    const int i0 = (int)pos2;
    const float t2 = pos2 - (float)i0;
    int i1 = i0 + 1; if (i1 > 511) i1 = 511;
    rows[s * 2 + 0] = i0; cw[s * 2 + 0] = tw[s] * (1.f - t2);
    rows[s * 2 + 1] = i1; cw[s * 2 + 1] = tw[s] * t2;
  }
  float acc = 0.f;
#pragma unroll
  for (int s = 0; s < 4; ++s) {
    const float* zr = z + ((size_t)b * HH2 + rows[s]) * 25;
    float a2 = 0.f;
    for (int w = 0; w < 25; ++w) a2 += zr[w] * Wconv2[w];
    acc += cw[s] * a2;
  }
  out[tid] = 1.0f / (1.0f + expf(-acc));
}

extern "C" void kernel_launch(void* const* d_in, const int* in_sizes, int n_in,
                              void* d_out, int out_size, void* d_ws, size_t ws_size,
                              hipStream_t stream) {
  (void)in_sizes; (void)n_in; (void)out_size; (void)ws_size;
  const float* p2r    = (const float*)d_in[0];
  const float* Wmsg   = (const float*)d_in[1];
  const float* gamma  = (const float*)d_in[2];
  const float* beta   = (const float*)d_in[3];
  const float* mean   = (const float*)d_in[4];
  const float* var    = (const float*)d_in[5];
  const float* Wup2   = (const float*)d_in[6];
  const float* Wconv  = (const float*)d_in[7];
  const float* Wconv2 = (const float*)d_in[8];
  float* out = (float*)d_out;

  char* ws = (char*)d_ws;
  const size_t SX = (size_t)25 * NS * NC * sizeof(float);   // 52,428,800 B
  float* X1  = (float*)(ws + 0);
  float* X2  = (float*)(ws + SX);
  bf16*  Cout = (bf16*)(ws + 0);                // aliases X1 (dead by conv time)
  bf16*  U    = (bf16*)(ws + 2 * SX);           // (b,h2,w,c) bf16, 52.4 MB
  size_t off = 2 * SX + (size_t)NB * HH2 * NW * NC * 2;
  bf16*  Wc  = (bf16*)(ws + off);  off += (size_t)65536 * 2;
  bf16*  Wb  = (bf16*)(ws + off);  off += (size_t)256 * KC * 2;
  float* bnA = (float*)(ws + off); off += 1024;
  float* bnB = (float*)(ws + off); off += 1024;
  bf16*  cb0 = (bf16*)(ws + off);  off += (size_t)NS * NC * 2;
  bf16*  cb1 = (bf16*)(ws + off);  off += (size_t)NS * NC * 2;
  float* z   = (float*)(ws + off); off += (size_t)NCONV * 4;   // ~161 MB total

  prep_kernel<<<(256 * KC + 255) / 256, 256, 0, stream>>>(
      Wmsg, Wup2, gamma, beta, mean, var, Wc, Wb, bnA, bnB);

  bf16* cb[2] = {cb0, cb1};

  // scan 1 (forward): xi = p2_r[...,w=i]; result slot = 24-i
  scan_init_kernel<<<2048, 256, 0, stream>>>(p2r, 0, X1, cb[0], bnA, bnB);
  for (int i = 1; i < 25; ++i)
    scan_step_kernel<<<256, 256, 0, stream>>>(
        Wc, cb[(i - 1) & 1], p2r, X1, cb[i & 1], bnA, bnB, i, 24 - i, 0);

  // scan 2 (reverse): xi = X1 slot (24-i); result slot = 24-i
  scan_init_kernel<<<2048, 256, 0, stream>>>(X1, 1, X2, cb[0], bnA, bnB);
  for (int i = 1; i < 25; ++i)
    scan_step_kernel<<<256, 256, 0, stream>>>(
        Wc, cb[(i - 1) & 1], X1, X2, cb[i & 1], bnA, bnB, 24 - i, 24 - i, 1);

  upsample_bf16_kernel<<<(NB * HH2 * NW * NC) / 256, 256, 0, stream>>>(X2, U);
  conv3_kernel<<<NCONV / 64, 256, 0, stream>>>(Wb, U, Cout);     // 1600 blocks
  chan_reduce_kernel<<<NCONV * 32 / 256, 256, 0, stream>>>(Cout, Wconv, z);
  tail_kernel<<<64, 256, 0, stream>>>(z, Wconv2, out);
}
